// AttentionPairBias_30588757082753
// MI455X (gfx1250) — matrix operand order repaired and checked by reading, not compiled, not run on hardware
//
#include <hip/hip_runtime.h>
#include <hip/hip_bf16.h>

// ---------------------------------------------------------------------------
// AttentionPairBias for MI455X (gfx1250, wave32).
// Dominant cost: streaming z (512 MB) once -> HBM-bound (~22us @ 23.3 TB/s).
// Pair-bias projection runs on v_wmma_f32_16x16x32_bf16 so compute never
// bottlenecks the stream; attention math uses v_wmma_f32_16x16x4_f32 (K=24).
// z is read with non-temporal hints so the single-use 512MB stream does not
// evict the L2-resident zb (32MB) / QKVG (6MB) working set that the attention
// kernel re-reads with scattered accesses.
// ---------------------------------------------------------------------------

typedef __bf16 bf16;
typedef __attribute__((ext_vector_type(16))) __bf16 v16bf;
typedef __attribute__((ext_vector_type(8)))  __bf16 v8bf;
typedef __attribute__((ext_vector_type(8)))  float  v8f;
typedef __attribute__((ext_vector_type(4)))  float  v4f;
typedef __attribute__((ext_vector_type(2)))  float  v2f;

#define NN   1024   // sequence length N
#define CQ   384    // channels of a
#define CZ   128    // channels of z
#define NHED 16     // heads
#define HDIM 24     // head dim

// ---- wave32 16-lane-group reductions (C-layout rows live in lane halves) --
__device__ __forceinline__ float redmax16(float x) {
#pragma unroll
  for (int m = 1; m < 16; m <<= 1) x = fmaxf(x, __shfl_xor(x, m, 32));
  return x;
}
__device__ __forceinline__ float redsum16(float x) {
#pragma unroll
  for (int m = 1; m < 16; m <<= 1) x += __shfl_xor(x, m, 32);
  return x;
}

__device__ __forceinline__ v16bf comb8(v8bf lo, v8bf hi) {
  v16bf r;
#pragma unroll
  for (int i = 0; i < 8; i++) { r[i] = lo[i]; r[i + 8] = hi[i]; }
  return r;
}

__device__ __forceinline__ v8f wmma_bf16(v16bf a, v16bf b, v8f c) {
  return __builtin_amdgcn_wmma_f32_16x16x32_bf16(false, a, false, b,
                                                 (short)0, c, false, false);
}
__device__ __forceinline__ v8f wmma_f32(v2f a, v2f b, v8f c) {
  return __builtin_amdgcn_wmma_f32_16x16x4_f32(false, a, false, b,
                                               (short)0, c, false, false);
}

// A operand (16x32 bf16, row-major source): lane m<16 holds K {s..s+7, 16+s..}
__device__ __forceinline__ v16bf load_a_tile(const bf16* A, int lda, int row0,
                                             int k0, int lane) {
  int m = lane & 15, s = (lane >> 4) * 8;
  const bf16* p = A + (size_t)(row0 + m) * lda + k0 + s;
  v8bf lo = *(const v8bf*)p;
  v8bf hi = *(const v8bf*)(p + 16);
  return comb8(lo, hi);
}
// B operand (32x16 bf16) from B^T row-major: lane n<16 holds K {s..s+7, 16+s..}
__device__ __forceinline__ v16bf load_b_tile(const bf16* BT, int ldb, int col0,
                                             int k0, int lane) {
  int n = lane & 15, s = (lane >> 4) * 8;
  const bf16* p = BT + (size_t)(col0 + n) * ldb + k0 + s;
  return comb8(*(const v8bf*)p, *(const v8bf*)(p + 16));
}

// ---------------------------------------------------------------------------
// Kernel 0: convert/transpose weights to bf16 (B^T layouts for WMMA).
// ---------------------------------------------------------------------------
__global__ void prep_kernel(const float* __restrict__ wq, const float* __restrict__ wk,
                            const float* __restrict__ wv, const float* __restrict__ wg,
                            const float* __restrict__ wo, const float* __restrict__ wz,
                            bf16* __restrict__ wT4, bf16* __restrict__ woT,
                            bf16* __restrict__ wzT) {
  const int M2 = CQ * CQ;
  int idx = blockIdx.x * 256 + threadIdx.x;
  if (idx < 4 * M2) {
    int id = idx / M2, r = idx % M2, o = r / CQ, i = r % CQ;
    const float* w = (id == 0) ? wq : (id == 1) ? wk : (id == 2) ? wv : wg;
    wT4[idx] = (bf16)w[(size_t)i * CQ + o];
  } else if (idx < 5 * M2) {
    int r = idx - 4 * M2, o = r / CQ, i = r % CQ;
    woT[r] = (bf16)wo[(size_t)i * CQ + o];
  } else if (idx < 5 * M2 + NHED * CZ) {
    int r = idx - 5 * M2, h = r / CZ, c = r % CZ;
    wzT[r] = (bf16)wz[(size_t)c * NHED + h];
  }
}

// ---------------------------------------------------------------------------
// Kernel 1: LayerNorm(a) -> bf16 [N][CQ]
// ---------------------------------------------------------------------------
__global__ void ln_a_kernel(const float* __restrict__ a, const float* __restrict__ w,
                            const float* __restrict__ b, bf16* __restrict__ aln) {
  int row = blockIdx.x, t = threadIdx.x;
  __shared__ float red[8];
  const float* ar = a + (size_t)row * CQ;
  float x0 = ar[t], x1 = ar[t + 128], x2 = ar[t + 256];
  float s = x0 + x1 + x2, s2 = x0 * x0 + x1 * x1 + x2 * x2;
#pragma unroll
  for (int m = 1; m < 32; m <<= 1) { s += __shfl_xor(s, m, 32); s2 += __shfl_xor(s2, m, 32); }
  if ((t & 31) == 0) { red[t >> 5] = s; red[4 + (t >> 5)] = s2; }
  __syncthreads();
  s  = red[0] + red[1] + red[2] + red[3];
  s2 = red[4] + red[5] + red[6] + red[7];
  float mu = s / CQ, var = s2 / CQ - mu * mu, rs = rsqrtf(var + 1e-5f);
  bf16* o = aln + (size_t)row * CQ;
  o[t]       = (bf16)((x0 - mu) * rs * w[t] + b[t]);
  o[t + 128] = (bf16)((x1 - mu) * rs * w[t + 128] + b[t + 128]);
  o[t + 256] = (bf16)((x2 - mu) * rs * w[t + 256] + b[t + 256]);
}

// ---------------------------------------------------------------------------
// Kernel 2: Q/K/V/G projections.  grid=(N/16, CQ/16, 4), one wave per tile.
// Q,K,V stored head-major [H][N][24] fp32; G = sigmoid(x+bg) fp32 [N][CQ].
// ---------------------------------------------------------------------------
__global__ void proj_kernel(const bf16* __restrict__ aln, const bf16* __restrict__ wT4,
                            const float* __restrict__ bg,
                            float* __restrict__ qb, float* __restrict__ kb,
                            float* __restrict__ vb, float* __restrict__ gb) {
  const int lane = threadIdx.x;
  const int row0 = blockIdx.x * 16, col0 = blockIdx.y * 16, id = blockIdx.z;
  const bf16* BT = wT4 + (size_t)id * CQ * CQ;
  v8f acc = {};
#pragma unroll
  for (int k0 = 0; k0 < CQ; k0 += 32)
    acc = wmma_bf16(load_a_tile(aln, CQ, row0, k0, lane),
                    load_b_tile(BT, CQ, col0, k0, lane), acc);
  const int n = lane & 15, hi = lane >> 4;
  const float qs = 0.20412414523193154f;  // 24^-0.5
#pragma unroll
  for (int i = 0; i < 8; i++) {
    int row = row0 + i + hi * 8, col = col0 + n;
    float v = acc[i];
    if (id == 3) {
      gb[(size_t)row * CQ + col] = 1.f / (1.f + __expf(-(v + bg[col])));
    } else {
      int h = col / HDIM, d = col % HDIM;
      float* dst = (id == 0) ? qb : (id == 1) ? kb : vb;
      dst[(size_t)h * NN * HDIM + (size_t)row * HDIM + d] = (id == 0) ? v * qs : v;
    }
  }
}

// ---------------------------------------------------------------------------
// Kernel 3: pair bias  zb[q][k][h] = (LN(z[q,k,:]) @ w_z)  as bf16 [N][N][H].
// One wave per 16 flattened rows (fixed q, 16 consecutive k). Lane pair
// (m, m+16) holds row m's 128 channels in the exact bf16 A-operand layout.
// z loads are non-temporal: 512MB single-use stream must not evict the
// L2-resident zb / QKVG working set.
// ---------------------------------------------------------------------------
__global__ void pair_bias_kernel(const float* __restrict__ z,
                                 const float* __restrict__ lnw,
                                 const float* __restrict__ lnb,
                                 const bf16* __restrict__ wzT,
                                 bf16* __restrict__ zb) {
  const int lane = threadIdx.x;
  const size_t rowbase = (size_t)blockIdx.x * 16;
  const int m = lane & 15, base = (lane >> 4) * 8;
  const float* zr = z + (rowbase + m) * CZ;

  v4f ch[16];
  float s = 0.f, s2 = 0.f;
#pragma unroll
  for (int u = 0; u < 8; u++) {          // 8 chunks of 8 channels per lane
    v4f x0 = __builtin_nontemporal_load((const v4f*)(zr + base + 16 * u));
    v4f x1 = __builtin_nontemporal_load((const v4f*)(zr + base + 16 * u + 4));
    ch[2 * u] = x0; ch[2 * u + 1] = x1;
#pragma unroll
    for (int t = 0; t < 4; t++) { s += x0[t] + x1[t]; s2 += x0[t] * x0[t] + x1[t] * x1[t]; }
  }
  s  += __shfl_xor(s, 16, 32);           // combine lane pair -> full 128-ch row
  s2 += __shfl_xor(s2, 16, 32);
  float mu = s * (1.f / CZ);
  float var = s2 * (1.f / CZ) - mu * mu;
  float rs = rsqrtf(var + 1e-5f);
#pragma unroll
  for (int u = 0; u < 8; u++) {
    v4f w0 = *(const v4f*)(lnw + base + 16 * u);
    v4f w1 = *(const v4f*)(lnw + base + 16 * u + 4);
    v4f b0 = *(const v4f*)(lnb + base + 16 * u);
    v4f b1 = *(const v4f*)(lnb + base + 16 * u + 4);
#pragma unroll
    for (int t = 0; t < 4; t++) {
      ch[2 * u][t]     = (ch[2 * u][t] - mu) * rs * w0[t] + b0[t];
      ch[2 * u + 1][t] = (ch[2 * u + 1][t] - mu) * rs * w1[t] + b1[t];
    }
  }
  v8f acc = {};
#pragma unroll
  for (int j = 0; j < 4; j++) {          // K = 128 = 4 x 32
    v16bf a;
#pragma unroll
    for (int t = 0; t < 4; t++) {
      a[t]      = (bf16)ch[4 * j][t];
      a[4 + t]  = (bf16)ch[4 * j + 1][t];
      a[8 + t]  = (bf16)ch[4 * j + 2][t];
      a[12 + t] = (bf16)ch[4 * j + 3][t];
    }
    v16bf b = comb8(*(const v8bf*)(wzT + (lane & 15) * CZ + j * 32 + (lane >> 4) * 8),
                    *(const v8bf*)(wzT + (lane & 15) * CZ + j * 32 + 16 + (lane >> 4) * 8));
    acc = wmma_bf16(a, b, acc);
  }
  // LDS transpose so the global store is one coalesced 512B burst.
  __shared__ __attribute__((aligned(16))) bf16 tile[256];
  const int hcol = lane & 15;
#pragma unroll
  for (int i = 0; i < 8; i++) {
    int kl = i + (lane >> 4) * 8;
    tile[kl * 16 + hcol] = (bf16)acc[i];
  }
  __syncthreads();
  const size_t qi = rowbase >> 10, kb0 = rowbase & (NN - 1);
  bf16* dst = zb + (qi * NN + kb0) * NHED;
  *(v8bf*)(dst + lane * 8) = *(const v8bf*)(tile + lane * 8);
}

// ---------------------------------------------------------------------------
// Kernel 4: flash attention w/ pair bias. grid=(N/16, H), one wave per block.
// ---------------------------------------------------------------------------
__global__ void attn_kernel(const float* __restrict__ qb, const float* __restrict__ kbuf,
                            const float* __restrict__ vbuf, const bf16* __restrict__ zb,
                            const float* __restrict__ mask, const float* __restrict__ gb,
                            bf16* __restrict__ og) {
  const int lane = threadIdx.x;
  const int q0 = blockIdx.x * 16, h = blockIdx.y;
  const int n = lane & 15, hi = lane >> 4;
  __shared__ __attribute__((aligned(16))) float ldsP[256];

  v2f qa[6];
  const float* qrow = qb + (size_t)h * NN * HDIM + (size_t)(q0 + n) * HDIM + hi * 2;
#pragma unroll
  for (int j = 0; j < 6; j++) qa[j] = *(const v2f*)(qrow + j * 4);

  float mrow[8], lrow[8];
#pragma unroll
  for (int i = 0; i < 8; i++) { mrow[i] = -1e30f; lrow[i] = 0.f; }
  v8f o0 = {}, o1 = {};

  for (int kt = 0; kt < NN / 16; kt++) {
    // S = Q K^T (16x16, K=24 via 6x wmma f32 16x16x4)
    v8f S = {};
    const float* krow = kbuf + (size_t)h * NN * HDIM + (size_t)(kt * 16 + n) * HDIM + hi * 2;
#pragma unroll
    for (int j = 0; j < 6; j++) S = wmma_f32(qa[j], *(const v2f*)(krow + j * 4), S);

    int col = kt * 16 + n;
    float mb = 1e9f * (mask[col] - 1.f);
    float p[8];
#pragma unroll
    for (int i = 0; i < 8; i++) {
      int row = q0 + i + hi * 8;
      float sv = S[i] + mb + (float)zb[((size_t)row * NN + col) * NHED + h];
      float tm = redmax16(sv);
      float nm = fmaxf(mrow[i], tm);
      float sc = __expf(mrow[i] - nm);
      float pv = __expf(sv - nm);
      lrow[i] = lrow[i] * sc + redsum16(pv);
      mrow[i] = nm;
      o0[i] *= sc; o1[i] *= sc;
      p[i] = pv;
    }
    __syncthreads();                    // prev-iter LDS reads done
#pragma unroll
    for (int i = 0; i < 8; i++) ldsP[(i + hi * 8) * 16 + n] = p[i];
    __syncthreads();
    // O += P @ V  (M=16 q, K=16 k as 4x4, N=24 d split 16+8 zero-padded)
    const float* vb0 = vbuf + (size_t)h * NN * HDIM + (size_t)kt * 16 * HDIM;
#pragma unroll
    for (int kk = 0; kk < 4; kk++) {
      v2f pa = *(const v2f*)(ldsP + n * 16 + kk * 4 + hi * 2);
      int r = kk * 4 + hi * 2;
      v2f b0 = { vb0[r * HDIM + n], vb0[(r + 1) * HDIM + n] };
      v2f b1 = { 0.f, 0.f };
      if (n < 8) { b1[0] = vb0[r * HDIM + 16 + n]; b1[1] = vb0[(r + 1) * HDIM + 16 + n]; }
      o0 = wmma_f32(pa, b0, o0);
      o1 = wmma_f32(pa, b1, o1);
    }
  }
#pragma unroll
  for (int i = 0; i < 8; i++) {
    int row = q0 + i + hi * 8;
    float inv = (lrow[i] > 0.f) ? 1.f / lrow[i] : 0.f;
    int c0 = h * HDIM + n;
    og[(size_t)row * CQ + c0] = (bf16)(o0[i] * inv * gb[(size_t)row * CQ + c0]);
    if (n < 8) {
      int c1 = h * HDIM + 16 + n;
      og[(size_t)row * CQ + c1] = (bf16)(o1[i] * inv * gb[(size_t)row * CQ + c1]);
    }
  }
}

// ---------------------------------------------------------------------------
// Kernel 5: out = og @ wo + bo   (bf16 WMMA, fp32 out, NT stores)
// ---------------------------------------------------------------------------
__global__ void outproj_kernel(const bf16* __restrict__ og, const bf16* __restrict__ woT,
                               const float* __restrict__ bo, float* __restrict__ out) {
  const int lane = threadIdx.x;
  const int row0 = blockIdx.x * 16, col0 = blockIdx.y * 16;
  v8f acc = {};
#pragma unroll
  for (int k0 = 0; k0 < CQ; k0 += 32)
    acc = wmma_bf16(load_a_tile(og, CQ, row0, k0, lane),
                    load_b_tile(woT, CQ, col0, k0, lane), acc);
  const int n = lane & 15, hi = lane >> 4;
#pragma unroll
  for (int i = 0; i < 8; i++) {
    int row = row0 + i + hi * 8, col = col0 + n;
    __builtin_nontemporal_store(acc[i] + bo[col], out + (size_t)row * CQ + col);
  }
}

// ---------------------------------------------------------------------------
extern "C" void kernel_launch(void* const* d_in, const int* in_sizes, int n_in,
                              void* d_out, int out_size, void* d_ws, size_t ws_size,
                              hipStream_t stream) {
  (void)in_sizes; (void)n_in; (void)out_size; (void)ws_size;
  const float* a    = (const float*)d_in[0];
  const float* z    = (const float*)d_in[1];
  const float* mask = (const float*)d_in[2];
  const float* lnaw = (const float*)d_in[3];
  const float* lnab = (const float*)d_in[4];
  const float* lnzw = (const float*)d_in[5];
  const float* lnzb = (const float*)d_in[6];
  const float* wz   = (const float*)d_in[7];
  const float* wq   = (const float*)d_in[8];
  const float* wk   = (const float*)d_in[9];
  const float* wv   = (const float*)d_in[10];
  const float* wg   = (const float*)d_in[11];
  const float* bg   = (const float*)d_in[12];
  const float* wo   = (const float*)d_in[13];
  const float* bo   = (const float*)d_in[14];
  float* out = (float*)d_out;

  char* p = (char*)d_ws;
  auto carve = [&](size_t bytes) -> char* {
    char* r = p; p += (bytes + 255) & ~(size_t)255; return r;
  };
  bf16*  aln = (bf16*)carve((size_t)NN * CQ * 2);
  bf16*  wT4 = (bf16*)carve((size_t)4 * CQ * CQ * 2);
  bf16*  woT = (bf16*)carve((size_t)CQ * CQ * 2);
  bf16*  wzT = (bf16*)carve((size_t)NHED * CZ * 2);
  float* qb  = (float*)carve((size_t)NN * CQ * 4);
  float* kb  = (float*)carve((size_t)NN * CQ * 4);
  float* vb  = (float*)carve((size_t)NN * CQ * 4);
  float* gb  = (float*)carve((size_t)NN * CQ * 4);
  bf16*  og  = (bf16*)carve((size_t)NN * CQ * 2);
  bf16*  zb  = (bf16*)carve((size_t)NN * NN * NHED * 2);   // 32 MB

  prep_kernel<<<(5 * CQ * CQ + NHED * CZ + 255) / 256, 256, 0, stream>>>(
      wq, wk, wv, wg, wo, wz, wT4, woT, wzT);
  ln_a_kernel<<<NN, 128, 0, stream>>>(a, lnaw, lnab, aln);
  proj_kernel<<<dim3(NN / 16, CQ / 16, 4), 32, 0, stream>>>(aln, wT4, bg, qb, kb, vb, gb);
  pair_bias_kernel<<<(NN * NN) / 16, 32, 0, stream>>>(z, lnzw, lnzb, wzT, zb);
  attn_kernel<<<dim3(NN / 16, NHED), 32, 0, stream>>>(qb, kb, vb, zb, mask, gb, og);
  outproj_kernel<<<dim3(NN / 16, CQ / 16), 32, 0, stream>>>(og, woT, bo, out);
}
